// MyModel_81166291960408
// MI455X (gfx1250) — matrix operand order repaired and checked by reading, not compiled, not run on hardware
//
#include <hip/hip_runtime.h>
#include <math.h>

typedef __attribute__((ext_vector_type(16))) _Float16 v16h;
typedef __attribute__((ext_vector_type(8)))  _Float16 v8h;
typedef __attribute__((ext_vector_type(8)))  float    v8f;

#define IN_DIM 5
#define H1 26
#define H2 121
#define T_STEPS 65536

// LSTM1: K = 5+26 = 31 -> padded 32 (1 chunk),  N = 104 -> 7 tiles of 16
// LSTM2: K = 26+121 = 147 -> padded 160 (5 chunks of 32), N = 484 -> 31 tiles of 16
constexpr int N1_TILES  = 7;
constexpr int N2_TILES  = 31;
constexpr int K2_CHUNKS = 5;

struct Smem {
  _Float16 W2f[N2_TILES * K2_CHUNKS * 512]; // WMMA B fragments, pre-swizzled  (155 KB)
  _Float16 W1f[N1_TILES * 512];             // WMMA B fragments                (7 KB)
  _Float16 u16[32];     // layer1 input vec: [x_t(5), h1(26), pad]
  _Float16 v16[160];    // layer2 input vec: [h1(26), h2(121), pad]
  float g1[112];        // layer1 raw gates (WMMA out)
  float g2[496];        // layer2 raw gates (WMMA out)
  float b1[112];        // b_ih1 + b_hh1
  float b2[496];        // b_ih2 + b_hh2
  float c1[32];
  float c2[128];
  float h2f[128];       // fp32 h2 for the linear head
  float wl[128];        // W_lin
  float xbuf[2][8];     // double-buffered x_t prefetch
};

// Branch-free transcendentals on the recurrence critical path:
// one v_exp_f32 (hw exp2) + one v_rcp_f32 each, correct saturation at +/-inf.
#define LOG2E 1.4426950408889634f
__device__ __forceinline__ float sigm(float x) {
  // 1 / (1 + 2^(-x*log2e)); x->-inf: exp2->inf, rcp->0; x->+inf: exp2->0, ->1
  return __builtin_amdgcn_rcpf(1.0f + __builtin_amdgcn_exp2f(-LOG2E * x));
}
__device__ __forceinline__ float tanh_fast(float x) {
  // 1 - 2/(1 + 2^(2x*log2e)); x->+inf -> 1, x->-inf -> -1
  return __builtin_fmaf(-2.0f,
                        __builtin_amdgcn_rcpf(1.0f + __builtin_amdgcn_exp2f(2.0f * LOG2E * x)),
                        1.0f);
}

__device__ __forceinline__ v16h cat16(v8h lo, v8h hi) {
  return __builtin_shufflevector(lo, hi, 0,1,2,3,4,5,6,7,8,9,10,11,12,13,14,15);
}

// B fragment (32x16 f16): lane-group 0 holds K=0..15 contiguous, group 1 K=16..31.
// Stored pre-swizzled: 16 contiguous halves per lane.
__device__ __forceinline__ v16h load_bfrag(const _Float16* p) {
  v8h lo = *(const v8h*)p;
  v8h hi = *(const v8h*)(p + 8);
  return cat16(lo, hi);
}

// A fragment (16x32 f16): lanes 0-15 hold K=0-15 contiguous,
// lanes 16-31 hold K=16-31 (matches B). All lanes mirror row 0 (matvec: rows 1-15
// replicate row 0; we only read D row 0).
__device__ __forceinline__ v16h load_afrag(const _Float16* vbase, int grp) {
  v8h lo = *(const v8h*)(vbase + grp * 16);
  v8h hi = *(const v8h*)(vbase + grp * 16 + 8);
  return cat16(lo, hi);
}

__global__ __launch_bounds__(1024, 1)
void lstm2_persistent_kernel(const float* __restrict__ x,
                             const float* __restrict__ W_ih1, const float* __restrict__ W_hh1,
                             const float* __restrict__ b_ih1, const float* __restrict__ b_hh1,
                             const float* __restrict__ W_ih2, const float* __restrict__ W_hh2,
                             const float* __restrict__ b_ih2, const float* __restrict__ b_hh2,
                             const float* __restrict__ W_lin, const float* __restrict__ b_lin,
                             const float* __restrict__ h1_0, const float* __restrict__ c1_0,
                             const float* __restrict__ h2_0, const float* __restrict__ c2_0,
                             float* __restrict__ out) {
  extern __shared__ char smem_raw[];
  Smem& S = *reinterpret_cast<Smem*>(smem_raw);

  const int tid  = threadIdx.x;
  const int wave = tid >> 5;
  const int lane = tid & 31;

  // ---------------- one-time init: swizzle weights into WMMA B-fragment layout ----------------
  // W2cat[n][k]: k<26 -> W_ih2[n][k], else W_hh2[n][k-26]; n<484, k<147, zero-padded.
  for (int e = tid; e < N2_TILES * K2_CHUNKS * 512; e += 1024) {
    int h    = e & 15;
    int l    = (e >> 4) & 31;
    int f    = e >> 9;
    int kc   = f % K2_CHUNKS;
    int tile = f / K2_CHUNKS;
    int n    = tile * 16 + (l & 15);
    int k    = kc * 32 + (l >> 4) * 16 + h;
    float v  = 0.0f;
    if (n < 4 * H2 && k < H1 + H2)
      v = (k < H1) ? W_ih2[n * H1 + k] : W_hh2[n * H2 + (k - H1)];
    S.W2f[e] = (_Float16)v;
  }
  // W1cat[n][k]: k<5 -> W_ih1[n][k], else W_hh1[n][k-5]; n<104, k<31.
  for (int e = tid; e < N1_TILES * 512; e += 1024) {
    int h    = e & 15;
    int l    = (e >> 4) & 31;
    int tile = e >> 9;
    int n    = tile * 16 + (l & 15);
    int k    = (l >> 4) * 16 + h;
    float v  = 0.0f;
    if (n < 4 * H1 && k < IN_DIM + H1)
      v = (k < IN_DIM) ? W_ih1[n * IN_DIM + k] : W_hh1[n * H1 + (k - IN_DIM)];
    S.W1f[e] = (_Float16)v;
  }
  for (int j = tid; j < 112; j += 1024) S.b1[j] = (j < 4 * H1) ? (b_ih1[j] + b_hh1[j]) : 0.f;
  for (int j = tid; j < 496; j += 1024) S.b2[j] = (j < 4 * H2) ? (b_ih2[j] + b_hh2[j]) : 0.f;
  if (tid < 32)  S.c1[tid]  = (tid < H1) ? c1_0[tid] : 0.f;
  if (tid < 128) S.c2[tid]  = (tid < H2) ? c2_0[tid] : 0.f;
  if (tid < 128) S.h2f[tid] = (tid < H2) ? h2_0[tid] : 0.f;
  if (tid < 128) S.wl[tid]  = (tid < H2) ? W_lin[tid] : 0.f;
  if (tid == 0)  S.u16[31]  = (_Float16)0.f;
  if (tid < IN_DIM) S.u16[tid] = (_Float16)x[tid];  // x_0 directly into layer1 input
  if (tid < H1) {
    _Float16 hh = (_Float16)h1_0[tid];
    S.u16[IN_DIM + tid] = hh;
    S.v16[tid]          = hh;
  }
  if (tid < 160 - H1) S.v16[H1 + tid] = (tid < H2) ? (_Float16)h2_0[tid] : (_Float16)0.f;
  const float y_bias = b_lin[0];
  __syncthreads();

  // ---------------- persistent sequential loop over T (4 barriers/step) ----------------
  for (int t = 0; t < T_STEPS; ++t) {
    // B: layer1 gates = W1 * [x_t; h1]  via WMMA (waves 0..6, one 16-col tile each).
    //    Wave 31 concurrently prefetches x_{t+1} from HBM into xbuf[(t+1)&1].
    if (wave < N1_TILES) {
      const int grp = lane >> 4;
      v16h a = load_afrag(S.u16, grp);
      v16h b = load_bfrag(&S.W1f[wave * 512 + lane * 16]);
      v8f  c = {};
      c = __builtin_amdgcn_wmma_f32_16x16x32_f16(false, a, false, b, (short)0, c, false, false);
      if (lane < 16) S.g1[(wave << 4) + lane] = c[0];  // D row 0
    } else if (wave == 31) {
      if (lane < IN_DIM && (t + 1) < T_STEPS)
        S.xbuf[(t + 1) & 1][lane] = x[(t + 1) * IN_DIM + lane];
    }
    __syncthreads();

    // C: layer1 cell update (threads 0..25); threads 26..30 stage x_{t+1} into u16
    //    (u16 was fully consumed by phase B of this step; barrier above orders both).
    if (tid < H1) {
      float ig = S.g1[tid]          + S.b1[tid];
      float fg = S.g1[H1 + tid]     + S.b1[H1 + tid];
      float gg = S.g1[2 * H1 + tid] + S.b1[2 * H1 + tid];
      float og = S.g1[3 * H1 + tid] + S.b1[3 * H1 + tid];
      float c  = sigm(fg) * S.c1[tid] + sigm(ig) * tanh_fast(gg);
      float h  = sigm(og) * tanh_fast(c);
      S.c1[tid] = c;
      _Float16 hh = (_Float16)h;
      S.u16[IN_DIM + tid] = hh;  // feeds layer1 next step
      S.v16[tid]          = hh;  // feeds layer2 this step
    } else if (tid < H1 + IN_DIM) {
      S.u16[tid - H1] = (_Float16)S.xbuf[(t + 1) & 1][tid - H1];  // x_{t+1} for next step
    }
    __syncthreads();

    // D: layer2 gates = W2 * [h1; h2] via WMMA (waves 0..30, one tile each, 5 K-chunks chained)
    if (wave < N2_TILES) {
      const int grp = lane >> 4;
      v16h a[K2_CHUNKS];
#pragma unroll
      for (int kc = 0; kc < K2_CHUNKS; ++kc) a[kc] = load_afrag(S.v16 + kc * 32, grp);
      v8f acc = {};
      const _Float16* wbase = &S.W2f[wave * (K2_CHUNKS * 512)];
#pragma unroll
      for (int kc = 0; kc < K2_CHUNKS; ++kc) {
        v16h b = load_bfrag(wbase + kc * 512 + lane * 16);
        acc = __builtin_amdgcn_wmma_f32_16x16x32_f16(false, a[kc], false, b, (short)0, acc,
                                                     false, false);
      }
      if (lane < 16) S.g2[(wave << 4) + lane] = acc[0];  // D row 0
    }
    __syncthreads();

    // E: layer2 cell update (fp32, branch-free fast transcendentals)
    if (tid < H2) {
      float ig = S.g2[tid]          + S.b2[tid];
      float fg = S.g2[H2 + tid]     + S.b2[H2 + tid];
      float gg = S.g2[2 * H2 + tid] + S.b2[2 * H2 + tid];
      float og = S.g2[3 * H2 + tid] + S.b2[3 * H2 + tid];
      float c  = sigm(fg) * S.c2[tid] + sigm(ig) * tanh_fast(gg);
      float h  = sigm(og) * tanh_fast(c);
      S.c2[tid]  = c;
      S.h2f[tid] = h;
      S.v16[H1 + tid] = (_Float16)h;  // feeds layer2 next step
    }
    __syncthreads();

    // F: linear head y_t = W_lin . h2 + b (wave 0; overlaps next iteration's phase B)
    if (wave == 0) {
      float s = 0.f;
      for (int j = lane; j < H2; j += 32) s += S.wl[j] * S.h2f[j];
      for (int off = 16; off > 0; off >>= 1) s += __shfl_down(s, off, 32);
      if (lane == 0) out[t] = s + y_bias;
    }
  }
}

extern "C" void kernel_launch(void* const* d_in, const int* in_sizes, int n_in,
                              void* d_out, int out_size, void* d_ws, size_t ws_size,
                              hipStream_t stream) {
  (void)in_sizes; (void)n_in; (void)d_ws; (void)ws_size; (void)out_size;
  const float* x     = (const float*)d_in[0];
  const float* W_ih1 = (const float*)d_in[1];
  const float* W_hh1 = (const float*)d_in[2];
  const float* b_ih1 = (const float*)d_in[3];
  const float* b_hh1 = (const float*)d_in[4];
  const float* W_ih2 = (const float*)d_in[5];
  const float* W_hh2 = (const float*)d_in[6];
  const float* b_ih2 = (const float*)d_in[7];
  const float* b_hh2 = (const float*)d_in[8];
  const float* W_lin = (const float*)d_in[9];
  const float* b_lin = (const float*)d_in[10];
  const float* h1_0  = (const float*)d_in[11];
  const float* c1_0  = (const float*)d_in[12];
  const float* h2_0  = (const float*)d_in[13];
  const float* c2_0  = (const float*)d_in[14];
  float* out = (float*)d_out;

  const size_t shmem = sizeof(Smem);  // ~173 KB < 320 KB/WGP on CDNA5
  lstm2_persistent_kernel<<<dim3(1), dim3(1024), shmem, stream>>>(
      x, W_ih1, W_hh1, b_ih1, b_hh1, W_ih2, W_hh2, b_ih2, b_hh2,
      W_lin, b_lin, h1_0, c1_0, h2_0, c2_0, out);
}
